// GATGraph_15822659519254
// MI455X (gfx1250) — matrix operand order repaired and checked by reading, not compiled, not run on hardware
//
#include <hip/hip_runtime.h>
#include <hip/hip_bf16.h>
#include <cstdint>

#define N_NODES 51200
#define N_GRAPHS 64
#define NODES_PER_GRAPH 800
#define N_EDGES 819200
#define DIM_IN 128
#define DIM_HIDDEN 64
#define HEADS 3
#define DIM_OUT 64

typedef __attribute__((ext_vector_type(16))) __bf16        v16bf;
typedef __attribute__((ext_vector_type(8)))  float         v8f;
typedef __attribute__((ext_vector_type(4)))  float         f32x4;
typedef __attribute__((ext_vector_type(4)))  int           i32x4;
typedef __attribute__((ext_vector_type(4)))  unsigned int  u32x4;
typedef __attribute__((ext_vector_type(8)))  unsigned int  u32x8;

// async-copy builtin takes pointers-to-v4i32 in explicit address spaces
typedef __attribute__((address_space(1))) i32x4* glob_v4i_ptr;
typedef __attribute__((address_space(3))) i32x4* lds_v4i_ptr;

#if defined(__has_builtin)
#if __has_builtin(__builtin_amdgcn_global_load_async_to_lds_b128)
#define ATHENA_HAVE_ASYNC_LDS 1
#endif
#if __has_builtin(__builtin_amdgcn_s_wait_asynccnt)
#define ATHENA_HAVE_WAIT_ASYNC 1
#endif
#endif

__device__ __forceinline__ float leakyf(float x, float s) { return x >= 0.f ? x : s * x; }

// Sign-aware float atomic max via integer atomics (init value must be -inf).
__device__ __forceinline__ void atomicMaxFloat(float* addr, float v) {
  if (v >= 0.f) atomicMax((int*)addr, __float_as_int(v));
  else          atomicMin((unsigned int*)addr, __float_as_uint(v));
}

// ---------------------------------------------------------------------------
// TDM: DMA a 2D f32 panel (rows x rowElems, row stride strideElems) from
// global memory into LDS at byte offset ldsOff. D# built per CDNA5 ISA §8.
// Issued per-wave (EXEC ignored); complete via s_wait_tensorcnt.
// ---------------------------------------------------------------------------
__device__ __forceinline__ void tdm_load_panel_f32(const float* gsrc,
                                                   unsigned ldsOff,
                                                   unsigned rows,
                                                   unsigned rowElems,
                                                   unsigned long long strideElems)
{
  const unsigned long long ga = (unsigned long long)(uintptr_t)gsrc;
  u32x4 g0;
  g0[0] = 1u;                                                  // count=1, user desc
  g0[1] = ldsOff;                                              // lds_addr (bytes)
  g0[2] = (unsigned)(ga & 0xFFFFFFFFu);                        // global_addr[31:0]
  g0[3] = (unsigned)((ga >> 32) & 0x01FFFFFFu) | (2u << 30);   // addr[56:32] | type=2

  const unsigned long long q0 = (2ull << 16)                           // data_size=4B
                              | ((unsigned long long)(rowElems & 0xFFFFu) << 48);
  const unsigned long long q1 = (unsigned long long)((rowElems >> 16) & 0xFFFFu)
                              | ((unsigned long long)rows << 16)        // tensor_dim1
                              | ((unsigned long long)(rowElems & 0xFFFFu) << 48); // tile_dim0
  const unsigned long long q2 = (unsigned long long)(rows & 0xFFFFu)    // tile_dim1
                              | ((strideElems & 0xFFFFFFFFull) << 32);  // stride0[31:0]
  const unsigned long long q3 = (strideElems >> 32) & 0xFFFFull;        // stride0[47:32]
  u32x8 g1;
  g1[0] = (unsigned)q0; g1[1] = (unsigned)(q0 >> 32);
  g1[2] = (unsigned)q1; g1[3] = (unsigned)(q1 >> 32);
  g1[4] = (unsigned)q2; g1[5] = (unsigned)(q2 >> 32);
  g1[6] = (unsigned)q3; g1[7] = (unsigned)(q3 >> 32);

  asm volatile("tensor_load_to_lds %0, %1" :: "s"(g0), "s"(g1) : "memory");
}

__device__ __forceinline__ void wait_async_zero() {
#ifdef ATHENA_HAVE_WAIT_ASYNC
  __builtin_amdgcn_s_wait_asynccnt(0);
#else
  asm volatile("s_wait_asynccnt 0x0" ::: "memory");
#endif
}

// ---------------------------------------------------------------------------
// bf16 WMMA GEMM:  C[M,N] = epilogue(A[M,K] @ B[K,N] + bias),  K in {64,128}.
// blockDim = 128 (4 waves). Block computes a 16 x 64 output slab.
//  - B panel (K x 64, row stride N) staged in LDS by the Tensor Data Mover.
//  - A panel (16 x K) staged in LDS via async global->LDS b128 copies.
//  - WMMA fragments fed from LDS: A as 4x ds_load_b128, B column reads.
// Fragment maps per ISA 7.12.2 (16-bit A 16x32 / B 32x16 / f32 C 16x16).
// ---------------------------------------------------------------------------
__global__ void wmma_gemm_kernel(const float* __restrict__ A,
                                 const float* __restrict__ B,
                                 const float* __restrict__ bias,
                                 float* __restrict__ C,
                                 int M, int N, int K, float slope, int doLeaky)
{
  __shared__ float As[16 * 128];   // 8 KB  (16 rows x K)
  __shared__ float Bs[128 * 64];   // 32 KB (K rows x 64 cols)

  const int tid  = threadIdx.x;
  const int lane = tid & 31;
  const int wave = tid >> 5;
  const int half = lane >> 4;
  const int l16  = lane & 15;
  const int m0   = blockIdx.x * 16;
  const int nSlab = blockIdx.y;            // 64-column slab index

  // ---- stage B panel via TDM (one DMA per block, issued by wave 0) ----
  if (tid < 32) {
    tdm_load_panel_f32(B + nSlab * 64,
                       (unsigned)(uintptr_t)(void*)Bs,
                       (unsigned)K, 64u, (unsigned long long)N);
  }

  // ---- stage A panel via async global->LDS copies (16 B per lane) ----
  const int rowQuads = K >> 2;             // float4s per row
  for (int t = tid; t < 16 * rowQuads; t += 128) {
    const int row = t / rowQuads;
    const int q   = t - row * rowQuads;
    const float* gp = A + (size_t)(m0 + row) * K + q * 4;
    float* lp = &As[row * K + q * 4];
#ifdef ATHENA_HAVE_ASYNC_LDS
    __builtin_amdgcn_global_load_async_to_lds_b128(
        (glob_v4i_ptr)(i32x4*)const_cast<float*>(gp),
        (lds_v4i_ptr)(i32x4*)lp, 0, 0);
#else
    *(f32x4*)lp = *(const f32x4*)gp;
#endif
  }

  wait_async_zero();
  __builtin_amdgcn_s_wait_tensorcnt(0);
  __syncthreads();

  // ---- WMMA main loop, fragments fed from LDS ----
  const int bcol = wave * 16 + l16;        // column within 64-wide slab
  v8f acc = {};
  for (int kk = 0; kk < K; kk += 32) {
    const int abase = l16 * K + kk + half * 8;
    const f32x4 a0 = *(const f32x4*)&As[abase];        // elems 0..3
    const f32x4 a1 = *(const f32x4*)&As[abase + 4];    // elems 4..7
    const f32x4 a2 = *(const f32x4*)&As[abase + 16];   // elems 8..11
    const f32x4 a3 = *(const f32x4*)&As[abase + 20];   // elems 12..15
    v16bf a, b;
#pragma unroll
    for (int i = 0; i < 4; ++i) {
      a[i]      = (__bf16)a0[i];
      a[i + 4]  = (__bf16)a1[i];
      a[i + 8]  = (__bf16)a2[i];
      a[i + 12] = (__bf16)a3[i];
    }
#pragma unroll
    for (int i = 0; i < 16; ++i)
      b[i] = (__bf16)Bs[(kk + half * 8 + (i < 8 ? i : i + 8)) * 64 + bcol];
    acc = __builtin_amdgcn_wmma_f32_16x16x32_bf16(false, a, false, b,
                                                  (short)0, acc, false, false);
  }

  // ---- epilogue ----
  const int col = nSlab * 64 + bcol;
  const float bv = bias ? bias[col] : 0.f;
#pragma unroll
  for (int r = 0; r < 8; ++r) {
    const int row = m0 + half * 8 + r;
    float v = acc[r] + bv;
    if (doLeaky) v = leakyf(v, slope);
    C[(size_t)row * N + col] = v;
  }
}

// ---- init: agg = 0, m = -inf, denom = 0 -----------------------------------
__global__ void init_kernel(float* __restrict__ mbuf, float* __restrict__ denom,
                            float* __restrict__ agg)
{
  const int i = blockIdx.x * blockDim.x + threadIdx.x;
  if (i < N_NODES * HEADS * DIM_HIDDEN) agg[i] = 0.f;
  if (i < N_NODES * HEADS) { mbuf[i] = -__builtin_inff(); denom[i] = 0.f; }
}

// ---- weff[k,h] = sum_c W_dst[k, h*64+c] * att_dst[h,c] --------------------
__global__ void weff_kernel(const float* __restrict__ W_dst,
                            const float* __restrict__ att_dst,
                            float* __restrict__ weff)
{
  const int i = blockIdx.x * blockDim.x + threadIdx.x;   // k*HEADS + hd
  if (i >= DIM_HIDDEN * HEADS) return;
  const int k = i / HEADS, hd = i % HEADS;
  float s = 0.f;
  for (int c = 0; c < DIM_HIDDEN; ++c)
    s += W_dst[(size_t)k * (HEADS * DIM_HIDDEN) + hd * DIM_HIDDEN + c] *
         att_dst[hd * DIM_HIDDEN + c];
  weff[i] = s;
}

// ---- a_src[n,h] = <x_src[n,h,:], att_src[h,:]> ; a_dst[n,h] = <h[n,:], weff[:,h]>
__global__ void attn_coef_kernel(const float* __restrict__ xsrc,
                                 const float* __restrict__ hbuf,
                                 const float* __restrict__ att_src,
                                 const float* __restrict__ weff,
                                 float* __restrict__ a_src,
                                 float* __restrict__ a_dst)
{
  const int i = blockIdx.x * blockDim.x + threadIdx.x;   // n*HEADS + hd
  if (i >= N_NODES * HEADS) return;
  const int n = i / HEADS, hd = i % HEADS;
  float s1 = 0.f, s2 = 0.f;
  for (int c = 0; c < DIM_HIDDEN; ++c) {
    s1 += xsrc[(size_t)n * (HEADS * DIM_HIDDEN) + hd * DIM_HIDDEN + c] *
          att_src[hd * DIM_HIDDEN + c];
    s2 += hbuf[(size_t)n * DIM_HIDDEN + c] * weff[c * HEADS + hd];
  }
  a_src[i] = s1;
  a_dst[i] = s2;
}

// ---- pass 1: logits (into attn slot) + segment max ------------------------
__global__ void edge_logit_kernel(const int* __restrict__ src,
                                  const int* __restrict__ dst,
                                  const float* __restrict__ a_src,
                                  const float* __restrict__ a_dst,
                                  float* __restrict__ attn,
                                  float* __restrict__ mbuf)
{
  const int i = blockIdx.x * blockDim.x + threadIdx.x;   // e*HEADS + hd
  if (i >= N_EDGES * HEADS) return;
  const int e = i / HEADS, hd = i % HEADS;
  const int s = src[e], d = dst[e];
  const float lg = leakyf(a_src[s * HEADS + hd] + a_dst[d * HEADS + hd], 0.2f);
  attn[i] = lg;
  atomicMaxFloat(&mbuf[(size_t)d * HEADS + hd], lg);
}

// ---- pass 2: e = exp(logit - m[dst]) (in place) + segment sum -------------
__global__ void edge_exp_kernel(const int* __restrict__ dst,
                                const float* __restrict__ mbuf,
                                float* __restrict__ attn,
                                float* __restrict__ denom)
{
  const int i = blockIdx.x * blockDim.x + threadIdx.x;
  if (i >= N_EDGES * HEADS) return;
  const int e = i / HEADS, hd = i % HEADS;
  const int d = dst[e];
  const float ev = __expf(attn[i] - mbuf[(size_t)d * HEADS + hd]);
  attn[i] = ev;
  atomicAdd(&denom[(size_t)d * HEADS + hd], ev);
}

// ---- pass 3: attn normalize (final output) + message scatter. One wave/edge.
__global__ void edge_scatter_kernel(const int* __restrict__ src,
                                    const int* __restrict__ dst,
                                    const float* __restrict__ xsrc,
                                    const float* __restrict__ denom,
                                    float* __restrict__ attn,
                                    float* __restrict__ agg)
{
  const int wave = threadIdx.x >> 5;
  const int lane = threadIdx.x & 31;
  const int e = blockIdx.x * 8 + wave;
  if (e >= N_EDGES) return;
  const int s = src[e], d = dst[e];
  float av = 0.f;
  if (lane < HEADS) {
    av = attn[(size_t)e * HEADS + lane] / denom[(size_t)d * HEADS + lane];
    attn[(size_t)e * HEADS + lane] = av;          // final attention coefficient
  }
  const float a0 = __shfl(av, 0);
  const float a1 = __shfl(av, 1);
  const float a2 = __shfl(av, 2);
#pragma unroll
  for (int j = 0; j < 6; ++j) {
    const int ch = lane + j * 32;                 // 0..191
    const float w = (ch < 64) ? a0 : (ch < 128 ? a1 : a2);
    atomicAdd(&agg[(size_t)d * (HEADS * DIM_HIDDEN) + ch],
              xsrc[(size_t)s * (HEADS * DIM_HIDDEN) + ch] * w);
  }
}

// ---- head mean + bias, build leaky(concat(h, out)) ------------------------
__global__ void outcat_kernel(const float* __restrict__ agg,
                              const float* __restrict__ hbuf,
                              const float* __restrict__ gbias,
                              float* __restrict__ catL)
{
  const int i = blockIdx.x * blockDim.x + threadIdx.x;   // n*64 + c
  if (i >= N_NODES * DIM_HIDDEN) return;
  const int n = i >> 6, c = i & 63;
  const size_t base = (size_t)n * (HEADS * DIM_HIDDEN);
  const float o = (agg[base + c] + agg[base + 64 + c] + agg[base + 128 + c]) *
                  (1.f / 3.f) + gbias[c];
  catL[(size_t)n * 128 + c]      = leakyf(hbuf[i], 0.01f);
  catL[(size_t)n * 128 + 64 + c] = leakyf(o, 0.01f);
}

// ---- per-graph mean pool --------------------------------------------------
__global__ void pool_kernel(const float* __restrict__ y, float* __restrict__ pooled)
{
  const int i = blockIdx.x * blockDim.x + threadIdx.x;   // g*64 + c
  if (i >= N_GRAPHS * DIM_OUT) return;
  const int g = i >> 6, c = i & 63;
  const float* p = y + (size_t)g * NODES_PER_GRAPH * DIM_OUT + c;
  float s = 0.f;
  for (int k = 0; k < NODES_PER_GRAPH; ++k) s += p[(size_t)k * DIM_OUT];
  pooled[i] = s * (1.f / NODES_PER_GRAPH);
}

extern "C" void kernel_launch(void* const* d_in, const int* in_sizes, int n_in,
                              void* d_out, int out_size, void* d_ws, size_t ws_size,
                              hipStream_t stream)
{
  const float* X        = (const float*)d_in[0];
  const float* W1       = (const float*)d_in[1];
  const float* b1       = (const float*)d_in[2];
  const float* W_src    = (const float*)d_in[3];
  const float* W_dst    = (const float*)d_in[4];
  const float* att_src  = (const float*)d_in[5];
  const float* att_dst  = (const float*)d_in[6];
  const float* gat_bias = (const float*)d_in[7];
  const float* Wf       = (const float*)d_in[8];
  const float* bf       = (const float*)d_in[9];
  const int*   edges    = (const int*)d_in[10];
  // d_in[11] = ptr (uniform graphs; NODES_PER_GRAPH constant used instead)

  const int* src = edges;
  const int* dst = edges + N_EDGES;

  // d_out: pooled [64,64] then attn [E,3]
  float* pooled   = (float*)d_out;
  float* attn_out = pooled + N_GRAPHS * DIM_OUT;

  // workspace layout (floats); catL overlays xsrc, y overlays agg
  float* ws    = (float*)d_ws;
  float* hbuf  = ws;                                   // 51200*64
  float* xsrc  = hbuf + (size_t)N_NODES * DIM_HIDDEN;  // 51200*192
  float* a_src = xsrc + (size_t)N_NODES * HEADS * DIM_HIDDEN;
  float* a_dst = a_src + N_NODES * HEADS;
  float* mbuf  = a_dst + N_NODES * HEADS;
  float* denom = mbuf + N_NODES * HEADS;
  float* agg   = denom + N_NODES * HEADS;              // 51200*192
  float* weff  = agg + (size_t)N_NODES * HEADS * DIM_HIDDEN;  // 64*3
  float* catL  = xsrc;                                 // 51200*128 (reuse)
  float* ybuf  = agg;                                  // 51200*64  (reuse)

  // 0. init m / denom / agg
  init_kernel<<<(N_NODES * HEADS * DIM_HIDDEN + 255) / 256, 256, 0, stream>>>(
      mbuf, denom, agg);
  // 1. effective dst-attention matrix (64x3)
  weff_kernel<<<1, 256, 0, stream>>>(W_dst, att_dst, weff);
  // 2. h = leaky(X @ W1 + b1)
  wmma_gemm_kernel<<<dim3(N_NODES / 16, 1), 128, 0, stream>>>(
      X, W1, b1, hbuf, N_NODES, DIM_HIDDEN, DIM_IN, 0.01f, 1);
  // 3. x_src = h @ W_src
  wmma_gemm_kernel<<<dim3(N_NODES / 16, 3), 128, 0, stream>>>(
      hbuf, W_src, nullptr, xsrc, N_NODES, HEADS * DIM_HIDDEN, DIM_HIDDEN, 0.f, 0);
  // 4. per-node attention scalars
  attn_coef_kernel<<<(N_NODES * HEADS + 255) / 256, 256, 0, stream>>>(
      xsrc, hbuf, att_src, weff, a_src, a_dst);
  // 5-7. edge softmax + scatter (attn_out doubles as staging)
  edge_logit_kernel<<<(N_EDGES * HEADS + 255) / 256, 256, 0, stream>>>(
      src, dst, a_src, a_dst, attn_out, mbuf);
  edge_exp_kernel<<<(N_EDGES * HEADS + 255) / 256, 256, 0, stream>>>(
      dst, mbuf, attn_out, denom);
  edge_scatter_kernel<<<N_EDGES / 8, 256, 0, stream>>>(
      src, dst, xsrc, denom, attn_out, agg);
  // 8. head mean + concat + leaky
  outcat_kernel<<<(N_NODES * DIM_HIDDEN + 255) / 256, 256, 0, stream>>>(
      agg, hbuf, gat_bias, catL);
  // 9. y = catL @ Wf + bf
  wmma_gemm_kernel<<<dim3(N_NODES / 16, 1), 128, 0, stream>>>(
      catL, Wf, bf, ybuf, N_NODES, DIM_OUT, 2 * DIM_HIDDEN, 0.f, 0);
  // 10. per-graph mean pool
  pool_kernel<<<(N_GRAPHS * DIM_OUT + 255) / 256, 256, 0, stream>>>(ybuf, pooled);
}
